// CoAttention_31576599560314
// MI455X (gfx1250) — matrix operand order repaired and checked by reading, not compiled, not run on hardware
//
#include <hip/hip_runtime.h>

typedef __attribute__((ext_vector_type(16))) __bf16 v16bf;
typedef __attribute__((ext_vector_type(8)))  __bf16 v8bf;
typedef __attribute__((ext_vector_type(8)))  float  v8f;
typedef __attribute__((ext_vector_type(4)))  unsigned int u32x4;
typedef __attribute__((ext_vector_type(8)))  int        i32x8;
typedef __attribute__((ext_vector_type(4)))  int        i32x4;

#define C_DIM 192
#define N_DIM 4096
#define B_DIM 8

#if __has_builtin(__builtin_amdgcn_tensor_load_to_lds)
#define HAVE_TDM 1
#else
#define HAVE_TDM 0
#endif

// Load a 16-element bf16 A/B operand from two 16-byte-aligned 8-element chunks.
__device__ inline v16bf ld_2x8(const __bf16* p0, const __bf16* p1) {
  v8bf lo = *(const v8bf*)p0;
  v8bf hi = *(const v8bf*)p1;
  v16bf r;
#pragma unroll
  for (int i = 0; i < 8; ++i) { r[i] = lo[i]; r[i + 8] = hi[i]; }
  return r;
}

__device__ inline v8f wmma_bf16(v16bf a, v16bf b, v8f c) {
  return __builtin_amdgcn_wmma_f32_16x16x32_bf16(false, a, false, b, (short)0, c,
                                                 false, false);
}

// Generic LDS pointer -> 32-bit LDS byte offset (low 32 bits of aperture addr).
__device__ inline unsigned lds_off(const void* p) {
  return (unsigned)(size_t)p;
}

// Issue a 2-D TDM tensor load: tile_d1 rows x tile_d0 elements (bf16),
// row stride stride0 elements, optional LDS padding encoded in g1w0.
__device__ inline void tdm_load_2d(unsigned lds_addr, const void* gaddr,
                                   unsigned g1w0, unsigned tensor_d0,
                                   unsigned tensor_d1, unsigned tile_d0,
                                   unsigned tile_d1, unsigned stride0) {
#if HAVE_TDM
  unsigned long long ga = (unsigned long long)(size_t)gaddr;
  u32x4 g0;
  g0[0] = 1u;                                  // count=1, user descriptor
  g0[1] = lds_addr;                            // LDS byte address
  g0[2] = (unsigned)(ga & 0xFFFFFFFFu);        // global_addr[31:0]
  g0[3] = (unsigned)((ga >> 32) & 0x1FFFFFFu)  // global_addr[56:32]
        | (2u << 30);                          // type=2 ("image")
  i32x8 g1;
  g1[0] = (int)g1w0;                                     // data_size|pad cfg
  g1[1] = (int)((tensor_d0 & 0xFFFFu) << 16);            // tensor_dim0 lo
  g1[2] = (int)((tensor_d0 >> 16) | ((tensor_d1 & 0xFFFFu) << 16));
  g1[3] = (int)((tensor_d1 >> 16) | (tile_d0 << 16));    // tile_dim0
  g1[4] = (int)(tile_d1 & 0xFFFFu);                      // tile_dim1 (dim2=0)
  g1[5] = (int)stride0;                                  // dim0 stride lo
  g1[6] = 0;                                             // stride hi / dim1 stride
  g1[7] = 0;
  i32x4 g2 = {0, 0, 0, 0};
  i32x4 g3 = {0, 0, 0, 0};
#if defined(__clang_major__) && (__clang_major__ >= 23)
  i32x8 g4 = {0, 0, 0, 0, 0, 0, 0, 0};
  __builtin_amdgcn_tensor_load_to_lds(g0, g1, g2, g3, g4, 0);
#else
  __builtin_amdgcn_tensor_load_to_lds(g0, g1, g2, g3, 0);
#endif
#else
  (void)lds_addr; (void)gaddr; (void)g1w0; (void)tensor_d0; (void)tensor_d1;
  (void)tile_d0; (void)tile_d1; (void)stride0;
#endif
}

// ltile layout: row = 3 chunks of 128B data + 16B pad each => 432 B/row.
// (Bank stride 108 dwords/row -> 44 mod 64 per lane, conflict-free for 16 lanes.)
#define LROW_BYTES 432
__device__ inline const __bf16* lt_elem(const __bf16* base, int row, int c) {
  return (const __bf16*)((const char*)base + row * LROW_BYTES + (c >> 6) * 144 +
                         (c & 63) * 2);
}

// ---------------- conversion kernels ----------------
__global__ void convert_w_kernel(const float* __restrict__ Wl,
                                 const float* __restrict__ Wr,
                                 __bf16* __restrict__ WlB,
                                 __bf16* __restrict__ WrB) {
  int idx = blockIdx.x * blockDim.x + threadIdx.x;
  if (idx < C_DIM * C_DIM) {
    WlB[idx] = (__bf16)Wl[idx];
    WrB[idx] = (__bf16)Wr[idx];
  }
}

__global__ void convert_v_kernel(const float* __restrict__ v,
                                 __bf16* __restrict__ vb, int total) {
  int idx = (blockIdx.x * blockDim.x + threadIdx.x) * 4;
  if (idx + 3 < total) {
    float4 x = *(const float4*)(v + idx);
    vb[idx + 0] = (__bf16)x.x;
    vb[idx + 1] = (__bf16)x.y;
    vb[idx + 2] = (__bf16)x.z;
    vb[idx + 3] = (__bf16)x.w;
  }
}

// ---------------- projection: Lt/Rt[b][n][c] = (W @ k + bias)^T in bf16 ------
__global__ __launch_bounds__(128)
void coattn_proj_kernel(const float* __restrict__ key,
                        const __bf16* __restrict__ WlB,
                        const __bf16* __restrict__ WrB,
                        const float* __restrict__ bl,
                        const float* __restrict__ br,
                        __bf16* __restrict__ Lt, __bf16* __restrict__ Rt) {
  __shared__ __bf16 ktile[64 * 200];  // [n_local][c], padded rows
  const int b = blockIdx.y;
  const int n0 = blockIdx.x * 64;
  const int tid = threadIdx.x;
  const int lane = tid & 31;
  const int wave = tid >> 5;
  const bool lolane = (lane < 16);
  const int nloc = lane & 15;

  const float* kB = key + (size_t)b * C_DIM * N_DIM;

  // Stage k[0..191][n0..n0+63] f32 -> transposed bf16 ktile[n][c]
  for (int idx = tid; idx < C_DIM * 16; idx += 128) {
    int c = idx >> 4, j4 = idx & 15;
    float4 v = *(const float4*)(kB + (size_t)c * N_DIM + n0 + j4 * 4);
    ktile[(j4 * 4 + 0) * 200 + c] = (__bf16)v.x;
    ktile[(j4 * 4 + 1) * 200 + c] = (__bf16)v.y;
    ktile[(j4 * 4 + 2) * 200 + c] = (__bf16)v.z;
    ktile[(j4 * 4 + 3) * 200 + c] = (__bf16)v.w;
  }
  __syncthreads();

  const int nl = wave * 16 + nloc;
  const int n = n0 + nl;

  // B-operand tiles: lane<16 holds K=c0..c0+7,c0+16..23 of column n; lane>=16 c0+8..15,c0+24..31.
  v16bf Bk[6];
#pragma unroll
  for (int kc = 0; kc < 6; ++kc) {
    const __bf16* p = &ktile[nl * 200 + kc * 32 + (lolane ? 0 : 8)];
    Bk[kc] = ld_2x8(p, p + 16);
  }

  const int hi8 = lolane ? 0 : 8;
  const int arow = lolane ? lane : (lane - 16);
  __bf16* LtB = Lt + (size_t)b * N_DIM * C_DIM;
  __bf16* RtB = Rt + (size_t)b * N_DIM * C_DIM;

  for (int ot = 0; ot < 12; ++ot) {
    int o0 = ot * 16;
    v8f accL = {};
    v8f accR = {};
#pragma unroll
    for (int kc = 0; kc < 6; ++kc) {
      int c0 = kc * 32;
      const __bf16* pa = WlB + (size_t)(o0 + arow) * C_DIM + c0 + (lolane ? 0 : 8);
      const __bf16* pb = WrB + (size_t)(o0 + arow) * C_DIM + c0 + (lolane ? 0 : 8);
      v16bf aL = ld_2x8(pa, pa + 16);
      v16bf aR = ld_2x8(pb, pb + 16);
      accL = wmma_bf16(aL, Bk[kc], accL);
      accR = wmma_bf16(aR, Bk[kc], accR);
    }
    v8bf sl, sr;
#pragma unroll
    for (int r = 0; r < 8; ++r) {
      sl[r] = (__bf16)(accL[r] + bl[o0 + hi8 + r]);
      sr[r] = (__bf16)(accR[r] + br[o0 + hi8 + r]);
    }
    *(v8bf*)(LtB + (size_t)n * C_DIM + o0 + hi8) = sl;
    *(v8bf*)(RtB + (size_t)n * C_DIM + o0 + hi8) = sr;
  }
}

// ---------------- fused flash-style attention ----------------
// out[b][c][n] = sum_m V[c,m] * softmax_m( sum_c' L[c',m] R[c',n] )
__global__ __launch_bounds__(128)
void coattn_fused_kernel(const __bf16* __restrict__ Lt,
                         const __bf16* __restrict__ Rt,
                         const __bf16* __restrict__ Vb,
                         float* __restrict__ out) {
  // Double-buffered tiles. ltile rows use 144B chunks (128B data + 16B pad);
  // vtile rows are 80B (64B data + 16B pad). Both match the TDM pad encoding.
  __shared__ __bf16 ltile[2][32 * (LROW_BYTES / 2)];
  __shared__ __bf16 vtile[2][192 * 40];

  const int b = blockIdx.y;
  const int n0 = blockIdx.x * 64;
  const int tid = threadIdx.x;
  const int lane = tid & 31;
  const int wave = tid >> 5;
  const bool lolane = (lane < 16);
  const int nloc = lane & 15;
  const int n = n0 + wave * 16 + nloc;
  const int arow = lolane ? lane : (lane - 16);
  const int aoff = lolane ? 0 : 8;

  const __bf16* LtB = Lt + (size_t)b * N_DIM * C_DIM;
  const __bf16* RtB = Rt + (size_t)b * N_DIM * C_DIM;
  const __bf16* VbB = Vb + (size_t)b * C_DIM * N_DIM;

  // Resident B-operand tiles for S: key_r column n, 6 chunks of 32 channels.
  v16bf Br[6];
#pragma unroll
  for (int kc = 0; kc < 6; ++kc) {
    const __bf16* p = RtB + (size_t)n * C_DIM + kc * 32 + (lolane ? 0 : 8);
    Br[kc] = ld_2x8(p, p + 16);
  }

  v8f O[12];
#pragma unroll
  for (int t = 0; t < 12; ++t) O[t] = (v8f){};
  float Mst = -3.0e38f;
  float Zst = 0.0f;

  const int STEPS = N_DIM / 32;  // 128

#if HAVE_TDM
  // Prologue: wave 0 DMA-loads step 0 into buffer 0.
  if (wave == 0) {
    // Lt rows [0,32): 32 x 192 bf16, pad 16B per 128B => 144B chunks.
    tdm_load_2d(lds_off(&ltile[0][0]), LtB,
                (1u << 16) | (1u << 20) | (4u << 22) | (3u << 25),
                C_DIM, N_DIM, C_DIM, 32, C_DIM);
    // V columns [0,32) for 192 channels: rows 64B + 16B pad => 80B rows.
    tdm_load_2d(lds_off(&vtile[0][0]), VbB,
                (1u << 16) | (1u << 20) | (3u << 22) | (3u << 25),
                N_DIM, C_DIM, 32, C_DIM, N_DIM);
  }
#endif

  for (int i = 0; i < STEPS; ++i) {
    const int m0 = i * 32;
    const int cur = i & 1;
    const __bf16* lbuf = &ltile[cur][0];
    const __bf16* vbuf = &vtile[cur][0];

#if HAVE_TDM
    if (wave == 0) {
      if (i + 1 < STEPS) {
        // Prefetch next step into the other buffer (free since the barrier at
        // the end of step i-1), then wait for the current step's two loads.
        const int mn = m0 + 32;
        tdm_load_2d(lds_off(&ltile[1 - cur][0]), LtB + (size_t)mn * C_DIM,
                    (1u << 16) | (1u << 20) | (4u << 22) | (3u << 25),
                    C_DIM, N_DIM, C_DIM, 32, C_DIM);
        tdm_load_2d(lds_off(&vtile[1 - cur][0]), VbB + mn,
                    (1u << 16) | (1u << 20) | (3u << 22) | (3u << 25),
                    N_DIM, C_DIM, 32, C_DIM, N_DIM);
        __builtin_amdgcn_s_wait_tensorcnt(2);
      } else {
        __builtin_amdgcn_s_wait_tensorcnt(0);
      }
    }
    __syncthreads();
#else
    // Manual staging fallback (same LDS layout as the TDM path).
#pragma unroll
    for (int it = 0; it < 6; ++it) {
      int idx = it * 128 + tid;  // 0..767
      {
        int row = idx / 24, g = idx % 24;  // 32 rows x 24 16B groups
        uint4 v = *(const uint4*)(LtB + (size_t)(m0 + row) * C_DIM + g * 8);
        char* dst = (char*)&ltile[cur][0] + row * LROW_BYTES + (g >> 3) * 144 +
                    (g & 7) * 16;
        *(uint4*)dst = v;
      }
      {
        int row = idx >> 2, ch = idx & 3;  // 192 rows x 4 16B groups
        uint4 v = *(const uint4*)(VbB + (size_t)row * N_DIM + m0 + ch * 8);
        *(uint4*)((char*)&vtile[cur][0] + row * 80 + ch * 16) = v;
      }
    }
    __syncthreads();
#endif

    // S tiles: s0 -> rows m0..m0+15, s1 -> m0+16..m0+31 (16 columns n).
    v8f s0 = {};
    v8f s1 = {};
#pragma unroll
    for (int kc = 0; kc < 6; ++kc) {
      int c0 = kc * 32 + aoff;
      v16bf a0 = ld_2x8(lt_elem(lbuf, arow, c0), lt_elem(lbuf, arow, c0 + 16));
      v16bf a1 =
          ld_2x8(lt_elem(lbuf, 16 + arow, c0), lt_elem(lbuf, 16 + arow, c0 + 16));
      s0 = wmma_bf16(a0, Br[kc], s0);
      s1 = wmma_bf16(a1, Br[kc], s1);
    }

    // Online softmax: lanes l and l+16 hold the two m-halves of column n.
    float mloc = -3.0e38f;
#pragma unroll
    for (int r = 0; r < 8; ++r) {
      mloc = fmaxf(mloc, s0[r]);
      mloc = fmaxf(mloc, s1[r]);
    }
    float mNew = fmaxf(Mst, fmaxf(mloc, __shfl_xor(mloc, 16, 32)));
    float scale = __expf(Mst - mNew);
    float p0v[8], p1v[8];
    float rsum = 0.0f;
#pragma unroll
    for (int r = 0; r < 8; ++r) {
      p0v[r] = __expf(s0[r] - mNew);
      p1v[r] = __expf(s1[r] - mNew);
      rsum += p0v[r] + p1v[r];
    }
    rsum += __shfl_xor(rsum, 16, 32);
    Zst = Zst * scale + rsum;
    Mst = mNew;
#pragma unroll
    for (int t = 0; t < 12; ++t)
#pragma unroll
      for (int r = 0; r < 8; ++r) O[t][r] *= scale;

    // Build P as the second-GEMM B operand entirely in registers.
    v16bf b2;
#pragma unroll
    for (int r = 0; r < 8; ++r) {
      float q0 = __shfl_xor(p0v[r], 16, 32);
      float q1 = __shfl_xor(p1v[r], 16, 32);
      b2[r]     = (__bf16)(lolane ? p0v[r] : q1);
      b2[r + 8] = (__bf16)(lolane ? q0 : p1v[r]);
    }

    // O[c,n] += V[c, m0..m0+31] x P
#pragma unroll
    for (int t = 0; t < 12; ++t) {
      const __bf16* vp = vbuf + (t * 16 + arow) * 40 + (lolane ? 0 : 16);
      v16bf a2 = ld_2x8(vp, vp + 8);
      O[t] = wmma_bf16(a2, b2, O[t]);
    }
    __syncthreads();  // everyone done reading buf[cur] before it is overwritten
  }

  float invZ = 1.0f / Zst;
  float* outB = out + (size_t)b * C_DIM * N_DIM;
  const int hi8 = lolane ? 0 : 8;
#pragma unroll
  for (int t = 0; t < 12; ++t) {
#pragma unroll
    for (int r = 0; r < 8; ++r) {
      int c = t * 16 + hi8 + r;
      outB[(size_t)c * N_DIM + n] = O[t][r] * invZ;
    }
  }
}

extern "C" void kernel_launch(void* const* d_in, const int* in_sizes, int n_in,
                              void* d_out, int out_size, void* d_ws, size_t ws_size,
                              hipStream_t stream) {
  const float* key   = (const float*)d_in[0];
  const float* value = (const float*)d_in[1];
  const float* Wl    = (const float*)d_in[2];
  const float* bl    = (const float*)d_in[3];
  const float* Wr    = (const float*)d_in[4];
  const float* br    = (const float*)d_in[5];
  float* out = (float*)d_out;

  // Workspace (bf16): WlB, WrB (192x192 each), Lt, Rt, Vb (8x4096x192 each).
  __bf16* WlB = (__bf16*)d_ws;
  __bf16* WrB = WlB + (size_t)C_DIM * C_DIM;
  __bf16* LtP = WrB + (size_t)C_DIM * C_DIM;
  __bf16* RtP = LtP + (size_t)B_DIM * N_DIM * C_DIM;
  __bf16* VbP = RtP + (size_t)B_DIM * N_DIM * C_DIM;

  const int wtotal = C_DIM * C_DIM;
  convert_w_kernel<<<(wtotal + 255) / 256, 256, 0, stream>>>(Wl, Wr, WlB, WrB);

  const int vtotal = B_DIM * C_DIM * N_DIM;
  convert_v_kernel<<<(vtotal / 4 + 255) / 256, 256, 0, stream>>>(value, VbP, vtotal);

  dim3 grid(N_DIM / 64, B_DIM);
  coattn_proj_kernel<<<grid, 128, 0, stream>>>(key, WlB, WrB, bl, br, LtP, RtP);
  coattn_fused_kernel<<<grid, 128, 0, stream>>>(LtP, RtP, VbP, out);
}
